// Attention_28638841929811
// MI455X (gfx1250) — matrix operand order repaired and checked by reading, not compiled, not run on hardware
//
#include <hip/hip_runtime.h>
#include <hip/hip_bf16.h>

typedef __bf16 bf16;
typedef __attribute__((ext_vector_type(16))) __bf16 v16bf;
typedef __attribute__((ext_vector_type(8)))  float  v8f;
typedef __attribute__((ext_vector_type(4)))  unsigned u32x4;
typedef __attribute__((ext_vector_type(8)))  int      i32x8;
typedef __attribute__((ext_vector_type(4)))  int      i32x4;

union U8x {
    unsigned u[8];
    v16bf    v;
};

#define SEQ    2048
#define DMODEL 2048
#define NH     16
#define NKV    4
#define HDIM   128

#if __has_builtin(__builtin_amdgcn_tensor_load_to_lds) && __has_builtin(__builtin_amdgcn_s_wait_tensorcnt)
#define USE_TDM 1
#else
#define USE_TDM 0
#endif

// ---------------------------------------------------------------- cast
__global__ void cast_f32_bf16(const float* __restrict__ in, bf16* __restrict__ out, int n) {
    int i = blockIdx.x * blockDim.x + threadIdx.x;
    if (i < n) out[i] = (bf16)in[i];
}

// ---------------------------------------------------------------- GEMM: C[M,N] = A[M,K] * B[N,K]^T
// block = 256 threads (8 waves). Block tile 128(M) x 128(N), K sliced by 32,
// double-buffered LDS, register prefetch of next global slice, and
// double-buffered B fragments so ds_load latency overlaps v_wmma issue.
__global__ __launch_bounds__(256)
void gemm_bf16_nt(const bf16* __restrict__ A, const bf16* __restrict__ Bm,
                  float* __restrict__ Cf, bf16* __restrict__ Cb,
                  int M, int N, int K) {
    __shared__ unsigned sA[2][128 * 17];   // [128][34 bf16] padded -> 17 dwords/row
    __shared__ unsigned sB[2][128 * 17];

    const int tid  = threadIdx.x;
    const int lane = tid & 31, wave = tid >> 5;
    const int laneN = lane & 15, half = lane >> 4;
    const int m0 = blockIdx.y * 128, n0 = blockIdx.x * 128;
    const unsigned* A32 = (const unsigned*)A;
    const unsigned* B32 = (const unsigned*)Bm;

    v8f acc[8] = {};

    unsigned ra[8], rb[8];
    const int rrow = tid >> 4, rcd = tid & 15;
#pragma unroll
    for (int i = 0; i < 8; ++i) {
        int row = rrow + i * 16;
        ra[i] = A32[(size_t)(m0 + row) * (K / 2) + rcd];
        rb[i] = B32[(size_t)(n0 + row) * (K / 2) + rcd];
    }
#pragma unroll
    for (int i = 0; i < 8; ++i) {
        int row = rrow + i * 16;
        sA[0][row * 17 + rcd] = ra[i];
        sB[0][row * 17 + rcd] = rb[i];
    }
    __syncthreads();

    int cur = 0;
    for (int kk = 0; kk < K; kk += 32) {
        const int nxt = kk + 32;
        if (nxt < K) {
#pragma unroll
            for (int i = 0; i < 8; ++i) {
                int row = rrow + i * 16;
                const unsigned* pa = &A32[(size_t)(m0 + row) * (K / 2) + (nxt / 2) + rcd];
                const unsigned* pb = &B32[(size_t)(n0 + row) * (K / 2) + (nxt / 2) + rcd];
                ra[i] = *pa;
                rb[i] = *pb;
                __builtin_prefetch(pa + 16, 0, 1);
                __builtin_prefetch(pb + 16, 0, 1);
            }
        }

        // A fragment (16x32) for this wave, shared across its 8 N-tiles
        U8x ua;
#pragma unroll
        for (int j = 0; j < 8; ++j) {
            int k2  = (j < 4 ? 0 : 8) + half * 4 + (j & 3);
            int row = wave * 16 + laneN;
            ua.u[j] = sA[cur][row * 17 + k2];
        }

        // software-pipelined B fragments: load nt+1 while WMMA consumes nt
        U8x ub[2];
#pragma unroll
        for (int j = 0; j < 8; ++j)
            ub[0].u[j] = sB[cur][laneN * 17 + (j < 4 ? 0 : 8) + half * 4 + (j & 3)];
#pragma unroll
        for (int nt = 0; nt < 8; ++nt) {
            if (nt < 7) {
                int col = (nt + 1) * 16 + laneN;
#pragma unroll
                for (int j = 0; j < 8; ++j)
                    ub[(nt + 1) & 1].u[j] = sB[cur][col * 17 + (j < 4 ? 0 : 8) + half * 4 + (j & 3)];
            }
            acc[nt] = __builtin_amdgcn_wmma_f32_16x16x32_bf16(
                false, ua.v, false, ub[nt & 1].v, (short)0, acc[nt], false, false);
        }

        if (nxt < K) {
#pragma unroll
            for (int i = 0; i < 8; ++i) {
                int row = rrow + i * 16;
                sA[cur ^ 1][row * 17 + rcd] = ra[i];
                sB[cur ^ 1][row * 17 + rcd] = rb[i];
            }
        }
        __syncthreads();
        cur ^= 1;
    }

#pragma unroll
    for (int nt = 0; nt < 8; ++nt) {
#pragma unroll
        for (int r = 0; r < 8; ++r) {
            int row = m0 + wave * 16 + r + half * 8;
            int col = n0 + nt * 16 + laneN;
            float v = acc[nt][r];
            if (Cf) Cf[(size_t)row * N + col] = v;
            if (Cb) Cb[(size_t)row * N + col] = (bf16)v;
        }
    }
}

// ---------------------------------------------------------------- RoPE + transpose to [B, H, S, HD] bf16
__global__ void rope_transpose(const float* __restrict__ in, bf16* __restrict__ out,
                               int H, int doRope, int total) {
    int id = blockIdx.x * blockDim.x + threadIdx.x;
    if (id >= total) return;
    int j = id & 63;
    int h = (id >> 6) % H;
    int s = (id / (64 * H)) % SEQ;
    int b = id / (64 * H * SEQ);

    size_t ibase = (((size_t)(b * SEQ + s) * H + h) * HDIM) + 2 * j;
    float x0 = in[ibase], x1 = in[ibase + 1];
    float o0 = x0, o1 = x1;
    if (doRope) {
        float ang = (float)s * __expf(-(float)j * (9.210340371976184f / 64.0f));
        float sn, c;
        __sincosf(ang, &sn, &c);
        o0 = x0 * c - x1 * sn;
        o1 = x1 * c + x0 * sn;
    }
    size_t obase = ((((size_t)b * H + h) * SEQ) + s) * HDIM + 2 * j;
    out[obase]     = (bf16)o0;
    out[obase + 1] = (bf16)o1;
}

// ---------------------------------------------------------------- flash attention (causal, GQA)
__global__ __launch_bounds__(256)
void flash_attn(const bf16* __restrict__ Q, const bf16* __restrict__ Kt,
                const bf16* __restrict__ Vt, bf16* __restrict__ O) {
    __shared__ unsigned sK[32 * 65];        // [32 keys][130 bf16] padded (65 dwords/row)
    __shared__ bf16     sVt[128 * 34];      // transposed [hd][34 keys] padded
    __shared__ bf16     sP[8 * 16 * 34];    // per-wave P tile [16][34] padded

    const int tid  = threadIdx.x;
    const int lane = tid & 31, wave = tid >> 5;
    const int laneN = lane & 15, half = lane >> 4;
    const int qb = blockIdx.x, h = blockIdx.y, b = blockIdx.z;
    const int kvh   = h >> 2;
    const int qrow0 = qb * 128 + wave * 16;

    const unsigned* Q32 = (const unsigned*)Q;
    const unsigned* K32 = (const unsigned*)Kt;
    const unsigned* V32 = (const unsigned*)Vt;

    // Q fragments: 16x128 as four 16x32 A-layout fragments
    U8x aq[4];
    size_t qbase2 = (((size_t)(b * NH + h) * SEQ) + (qrow0 + laneN)) * (HDIM / 2);
#pragma unroll
    for (int c = 0; c < 4; ++c)
#pragma unroll
        for (int j = 0; j < 8; ++j) {
            int k2 = c * 16 + (j < 4 ? 0 : 8) + half * 4 + (j & 3);
            aq[c].u[j] = Q32[qbase2 + k2];
        }

    v8f acc[8] = {};
    float rmax[8], rsum[8];
#pragma unroll
    for (int r = 0; r < 8; ++r) { rmax[r] = -3.0e38f; rsum[r] = 0.f; }

    const size_t kvbase2 = ((size_t)(b * NKV + kvh) * SEQ) * (HDIM / 2);
    const int kend = qb * 128 + 128;

#if USE_TDM
    // D# group 1 invariants: data_size=2B, pad_enable,
    // pad_interval=64 dwords (code 5), pad_amount=1 dword (code 0)
    const unsigned g1d0 = (1u << 16) | (1u << 20) | (5u << 22);
    const unsigned ldsK = (unsigned)(size_t)(void*)&sK[0];   // flat low 32 bits = LDS offset
#endif

    for (int kc = 0; kc < kend; kc += 32) {
#if USE_TDM
        if (wave == 0) {
            unsigned long long ga =
                (unsigned long long)(size_t)(const void*)&Kt[kvbase2 * 2 + (size_t)kc * HDIM];
            u32x4 g0;
            g0.x = 1u;                                // count=1, user descriptor
            g0.y = ldsK;                              // lds_addr
            g0.z = (unsigned)(ga & 0xffffffffu);      // global_addr[31:0]
            g0.w = (unsigned)(ga >> 32) | (2u << 30); // global_addr[56:32] | type=2
            i32x8 g1;
            g1[0] = (int)g1d0;
            g1[1] = (int)(HDIM << 16);                // tensor_dim0 = 128
            g1[2] = (int)(32u << 16);                 // tensor_dim1 = 32
            g1[3] = (int)(HDIM << 16);                // tile_dim0 = 128
            g1[4] = 32;                               // tile_dim1 = 32
            g1[5] = HDIM;                             // tensor_dim0_stride = 128
            g1[6] = 0;
            g1[7] = 0;
            i32x4 z4 = {0, 0, 0, 0};
            i32x8 z8 = {0, 0, 0, 0, 0, 0, 0, 0};
            __builtin_amdgcn_tensor_load_to_lds(g0, g1, z4, z4, z8, 0);
        }
#else
#pragma unroll
        for (int i = 0; i < 8; ++i) {
            int d = tid + i * 256;
            int row = d >> 6, cd = d & 63;
            sK[row * 65 + cd] = K32[kvbase2 + (size_t)(kc + row) * (HDIM / 2) + cd];
        }
#endif
        // stage V transposed: sVt[hd][key]
#pragma unroll
        for (int i = 0; i < 8; ++i) {
            int d = tid + i * 256;
            int row = d >> 6, cd = d & 63;
            unsigned vv = V32[kvbase2 + (size_t)(kc + row) * (HDIM / 2) + cd];
            bf16 lo = __builtin_bit_cast(bf16, (unsigned short)(vv & 0xffffu));
            bf16 hi = __builtin_bit_cast(bf16, (unsigned short)(vv >> 16));
            sVt[(2 * cd)     * 34 + row] = lo;
            sVt[(2 * cd + 1) * 34 + row] = hi;
        }
#if USE_TDM
        if (wave == 0) __builtin_amdgcn_s_wait_tensorcnt(0);
#endif
        __syncthreads();

        if (kc <= qrow0 + 15) {                 // uniform per wave
            // S = Q * K^T : two 16x16 tiles, reduced over hd=128 (8 WMMA),
            // B fragments double-buffered against the WMMA issue.
            v8f sc[2] = {};
            U8x ub[2];
#pragma unroll
            for (int j = 0; j < 8; ++j)
                ub[0].u[j] = sK[laneN * 65 + (j < 4 ? 0 : 8) + half * 4 + (j & 3)];
#pragma unroll
            for (int f = 0; f < 8; ++f) {       // f = kt*4 + c
                if (f < 7) {
                    int kt1 = (f + 1) >> 2, c1 = (f + 1) & 3;
#pragma unroll
                    for (int j = 0; j < 8; ++j) {
                        int key = kt1 * 16 + laneN;
                        int k2  = c1 * 16 + (j < 4 ? 0 : 8) + half * 4 + (j & 3);
                        ub[(f + 1) & 1].u[j] = sK[key * 65 + k2];
                    }
                }
                int kt = f >> 2, c = f & 3;
                sc[kt] = __builtin_amdgcn_wmma_f32_16x16x32_bf16(
                    false, aq[c].v, false, ub[f & 1].v, (short)0, sc[kt], false, false);
            }

            const float scale = 0.08838834764831843f;   // 1/sqrt(128)
            bf16* myP = &sP[wave * 16 * 34];
#pragma unroll
            for (int r = 0; r < 8; ++r) {
                int mrel = r + half * 8;
                int qg = qrow0 + mrel;
                float s0 = sc[0][r] * scale;
                float s1 = sc[1][r] * scale;
                if (kc + laneN > qg)      s0 = -3.0e38f;
                if (kc + 16 + laneN > qg) s1 = -3.0e38f;
                float mx = fmaxf(s0, s1);
#pragma unroll
                for (int d = 1; d < 16; d <<= 1)
                    mx = fmaxf(mx, __shfl_xor(mx, d, 32));
                float mnew = fmaxf(rmax[r], mx);
                float corr = __expf(rmax[r] - mnew);
                rmax[r] = mnew;
                float e0 = __expf(s0 - mnew), e1 = __expf(s1 - mnew);
                float ps = e0 + e1;
#pragma unroll
                for (int d = 1; d < 16; d <<= 1)
                    ps += __shfl_xor(ps, d, 32);
                rsum[r] = rsum[r] * corr + ps;
#pragma unroll
                for (int t = 0; t < 8; ++t) acc[t][r] *= corr;
                myP[mrel * 34 + laneN]      = (bf16)e0;
                myP[mrel * 34 + 16 + laneN] = (bf16)e1;
            }
            asm volatile("s_wait_dscnt 0" ::: "memory");

            // P (16x32) in A-layout
            const unsigned* P32 = (const unsigned*)myP;
            U8x ap;
#pragma unroll
            for (int j = 0; j < 8; ++j) {
                int k2 = (j < 4 ? 0 : 8) + half * 4 + (j & 3);
                ap.u[j] = P32[laneN * 17 + k2];
            }
            // acc += P * V : 8 hd tiles, B fragments double-buffered
            const unsigned* sVt32 = (const unsigned*)sVt;
#pragma unroll
            for (int j = 0; j < 8; ++j)
                ub[0].u[j] = sVt32[laneN * 17 + (j < 4 ? 0 : 8) + half * 4 + (j & 3)];
#pragma unroll
            for (int t = 0; t < 8; ++t) {
                if (t < 7) {
#pragma unroll
                    for (int j = 0; j < 8; ++j) {
                        int hd = (t + 1) * 16 + laneN;
                        ub[(t + 1) & 1].u[j] = sVt32[hd * 17 + (j < 4 ? 0 : 8) + half * 4 + (j & 3)];
                    }
                }
                acc[t] = __builtin_amdgcn_wmma_f32_16x16x32_bf16(
                    false, ap.v, false, ub[t & 1].v, (short)0, acc[t], false, false);
            }
        }
        __syncthreads();
    }

    // write attn output as bf16 [B, S, NH*HD]
#pragma unroll
    for (int r = 0; r < 8; ++r) {
        int mrel = r + half * 8;
        float inv = 1.0f / rsum[r];
        size_t obase = ((size_t)(b * SEQ + qrow0 + mrel)) * DMODEL + h * HDIM;
#pragma unroll
        for (int t = 0; t < 8; ++t)
            O[obase + t * 16 + laneN] = (bf16)(acc[t][r] * inv);
    }
}

// ---------------------------------------------------------------- launch
extern "C" void kernel_launch(void* const* d_in, const int* in_sizes, int n_in,
                              void* d_out, int out_size, void* d_ws, size_t ws_size,
                              hipStream_t stream) {
    const float* x  = (const float*)d_in[0];
    const float* wq = (const float*)d_in[1];
    const float* wk = (const float*)d_in[2];
    const float* wv = (const float*)d_in[3];
    const float* wo = (const float*)d_in[4];
    float* out = (float*)d_out;

    const int Mrows = 2 * SEQ;   // 4096 token rows
    char* ws = (char*)d_ws;
    size_t off = 0;
    auto alloc = [&](size_t bytes) {
        void* p = ws + off;
        off += (bytes + 255) & ~(size_t)255;
        return p;
    };
    bf16*  xb  = (bf16*) alloc((size_t)Mrows * DMODEL * 2);
    bf16*  wqb = (bf16*) alloc((size_t)DMODEL * DMODEL * 2);
    bf16*  wkb = (bf16*) alloc((size_t)(NKV * HDIM) * DMODEL * 2);
    bf16*  wvb = (bf16*) alloc((size_t)(NKV * HDIM) * DMODEL * 2);
    bf16*  wob = (bf16*) alloc((size_t)DMODEL * DMODEL * 2);
    float* qf  = (float*)alloc((size_t)Mrows * DMODEL * 4);
    float* kf  = (float*)alloc((size_t)Mrows * (NKV * HDIM) * 4);
    float* vf  = (float*)alloc((size_t)Mrows * (NKV * HDIM) * 4);
    bf16*  qr  = (bf16*) alloc((size_t)Mrows * DMODEL * 2);
    bf16*  kr  = (bf16*) alloc((size_t)Mrows * (NKV * HDIM) * 2);
    bf16*  vr  = (bf16*) alloc((size_t)Mrows * (NKV * HDIM) * 2);
    bf16*  ab  = (bf16*) alloc((size_t)Mrows * DMODEL * 2);

    auto cast = [&](const float* i, bf16* o, size_t n) {
        cast_f32_bf16<<<(unsigned)((n + 255) / 256), 256, 0, stream>>>(i, o, (int)n);
    };
    cast(x,  xb,  (size_t)Mrows * DMODEL);
    cast(wq, wqb, (size_t)DMODEL * DMODEL);
    cast(wk, wkb, (size_t)(NKV * HDIM) * DMODEL);
    cast(wv, wvb, (size_t)(NKV * HDIM) * DMODEL);
    cast(wo, wob, (size_t)DMODEL * DMODEL);

    // projections: C = X * W^T
    gemm_bf16_nt<<<dim3(DMODEL / 128, Mrows / 128), 256, 0, stream>>>(
        xb, wqb, qf, nullptr, Mrows, DMODEL, DMODEL);
    gemm_bf16_nt<<<dim3((NKV * HDIM) / 128, Mrows / 128), 256, 0, stream>>>(
        xb, wkb, kf, nullptr, Mrows, NKV * HDIM, DMODEL);
    gemm_bf16_nt<<<dim3((NKV * HDIM) / 128, Mrows / 128), 256, 0, stream>>>(
        xb, wvb, vf, nullptr, Mrows, NKV * HDIM, DMODEL);

    // RoPE + [B,S,H,HD] -> [B,H,S,HD] bf16
    {
        int nq = Mrows * NH * (HDIM / 2);
        int nk = Mrows * NKV * (HDIM / 2);
        rope_transpose<<<(nq + 255) / 256, 256, 0, stream>>>(qf, qr, NH, 1, nq);
        rope_transpose<<<(nk + 255) / 256, 256, 0, stream>>>(kf, kr, NKV, 1, nk);
        rope_transpose<<<(nk + 255) / 256, 256, 0, stream>>>(vf, vr, NKV, 0, nk);
    }

    // causal GQA flash attention
    flash_attn<<<dim3(SEQ / 128, NH, 2), 256, 0, stream>>>(qr, kr, vr, ab);

    // output projection: out = attn * wo^T (fp32 result)
    gemm_bf16_nt<<<dim3(DMODEL / 128, Mrows / 128), 256, 0, stream>>>(
        ab, wob, out, nullptr, Mrows, DMODEL, DMODEL);
}
